// Elman_10239202034084
// MI455X (gfx1250) — matrix operand order repaired and checked by reading, not compiled, not run on hardware
//
#include <hip/hip_runtime.h>

// Elman RNN: B=256, T=512, E=300.
//   Phase 0:  H[b,t,:] = x[b,t,:] @ W1[:, :300]^T + b1               (parallel GEMM)
//   Phase 1:  H[b,t,:] = relu(H[b,t,:] + H[b,t-1,:] @ W1[:,300:]^T)  (512 sequential)
//   copy:     h_final = H[:, T-1, :]
//   Phase 2:  out[b,t,:] = H[b,t,:] @ W2^T + b2    (in-place over H = d_out)
// bf16 WMMA (v_wmma_f32_16x16x32_bf16), f32 accumulate.
// Boundary handling specialized at compile time: kb 0..8 / nt 0..17 are branch-free.

#define B_   256
#define T_   512
#define E_   300
#define E2_  600
#define KB_  10   // ceil(300/32) K-blocks (kb=9 is the K-tail)
#define NT_  19   // ceil(300/16) N-tiles (nt=18 is the N-tail)

typedef __attribute__((ext_vector_type(16))) __bf16 v16bf;
typedef __attribute__((ext_vector_type(8)))  float  v8f;

__device__ __forceinline__ v16bf cvt16(float4 u0, float4 u1, float4 u2, float4 u3) {
  v16bf v;
  v[0]=(__bf16)u0.x;  v[1]=(__bf16)u0.y;  v[2]=(__bf16)u0.z;  v[3]=(__bf16)u0.w;
  v[4]=(__bf16)u1.x;  v[5]=(__bf16)u1.y;  v[6]=(__bf16)u1.z;  v[7]=(__bf16)u1.w;
  v[8]=(__bf16)u2.x;  v[9]=(__bf16)u2.y;  v[10]=(__bf16)u2.z; v[11]=(__bf16)u2.w;
  v[12]=(__bf16)u3.x; v[13]=(__bf16)u3.y; v[14]=(__bf16)u3.z; v[15]=(__bf16)u3.w;
  return v;
}

// ---- A fragment: 16x32 bf16, rows m0..m0+15 of row-major [.. x lda] f32 matrix.
// Contiguous k order used here (same as B): lane L (half h=L>>4) holds row m0+(L&15), K = kb*32 + h*16 + {0..15}.
// Fast path valid for kb*32+32 <= Klim.
__device__ __forceinline__ v16bf load_A_full(const float* __restrict__ base, long lda,
                                             int m0, int kb, int lane) {
  const int row   = m0 + (lane & 15);
  const int khalf = lane >> 4;
  const float* rp = base + (long)row * lda + kb * 32 + khalf * 16;
  const float4* pa = (const float4*)rp;         // 16B aligned (rows are 1200B)
  const float4* pb = (const float4*)(rp + 8);
  return cvt16(pa[0], pa[1], pb[0], pb[1]);
}

__device__ __forceinline__ v16bf load_A_tail(const float* __restrict__ base, long lda,
                                             int m0, int kb, int Klim, int lane) {
  const int row   = m0 + (lane & 15);
  const int khalf = lane >> 4;
  const float* rp = base + (long)row * lda;
  const int k1 = kb * 32 + khalf * 16;
  const int k2 = k1 + 8;
  v16bf a;
#pragma unroll
  for (int i = 0; i < 8; ++i) {
    a[i]   = (k1 + i < Klim) ? (__bf16)rp[k1 + i] : (__bf16)0.f;
    a[i+8] = (k2 + i < Klim) ? (__bf16)rp[k2 + i] : (__bf16)0.f;
  }
  return a;
}

// ---- B fragment: 32x16 bf16 where B[k][n] = W[n*ldw + koff + k] (i.e. @ W^T).
// Contiguous k order used here (same as A): lane L holds col n0+(L&15), K = kb*32 + (L>>4)*16 + {0..15}.
// Fast path valid when all lanes' n < Nlim and kb*32+32 <= Klim.
__device__ __forceinline__ v16bf load_B_full(const float* __restrict__ W, int ldw,
                                             int koff, int n0, int kb, int lane) {
  const int n     = n0 + (lane & 15);
  const int khalf = lane >> 4;
  const float4* p = (const float4*)(W + (long)n * ldw + koff + kb * 32 + khalf * 16);
  return cvt16(p[0], p[1], p[2], p[3]);
}

// General predicated B load (K-tail and/or N-tail).
__device__ __forceinline__ v16bf load_B_pred(const float* __restrict__ W, int ldw,
                                             int koff, int n0, int kb,
                                             int Klim, int Nlim, int lane) {
  const int n     = n0 + (lane & 15);
  const int khalf = lane >> 4;
  const int kbase = kb * 32 + khalf * 16;
  v16bf b;
  if (n < Nlim && kbase + 16 <= Klim) {
    const float4* p = (const float4*)(W + (long)n * ldw + koff + kbase);
    b = cvt16(p[0], p[1], p[2], p[3]);
  } else if (n < Nlim) {
    const float* p = W + (long)n * ldw + koff;
#pragma unroll
    for (int i = 0; i < 16; ++i)
      b[i] = (kbase + i < Klim) ? (__bf16)p[kbase + i] : (__bf16)0.f;
  } else {
#pragma unroll
    for (int i = 0; i < 16; ++i) b[i] = (__bf16)0.f;
  }
  return b;
}

// One 16x16 output tile: 9 branch-free K-blocks + 1 predicated K-tail.
// NTAIL=true adds per-lane n<300 predication (only for nt==18).
template <bool NTAIL>
__device__ __forceinline__ v8f tile_accum(const v16bf* __restrict__ a,
                                          const float* __restrict__ W, int ldw,
                                          int koff, int n0, int lane) {
  v8f acc = {};
#pragma unroll
  for (int kb = 0; kb < KB_ - 1; ++kb) {
    v16bf bm = NTAIL ? load_B_pred(W, ldw, koff, n0, kb, E_, E_, lane)
                     : load_B_full(W, ldw, koff, n0, kb, lane);
    acc = __builtin_amdgcn_wmma_f32_16x16x32_bf16(false, a[kb], false, bm,
                                                  (short)0, acc, false, false);
  }
  v16bf bm = load_B_pred(W, ldw, koff, n0, KB_ - 1, E_, E_, lane);
  acc = __builtin_amdgcn_wmma_f32_16x16x32_bf16(false, a[KB_ - 1], false, bm,
                                                (short)0, acc, false, false);
  return acc;
}

// Load a full 16-row x 300-col A panel (10 fragments) into registers.
__device__ __forceinline__ void load_A_panel(v16bf* __restrict__ a,
                                             const float* __restrict__ base, long lda,
                                             int m0, int lane) {
#pragma unroll
  for (int kb = 0; kb < KB_ - 1; ++kb) a[kb] = load_A_full(base, lda, m0, kb, lane);
  a[KB_ - 1] = load_A_tail(base, lda, m0, KB_ - 1, E_, lane);
}

// Phase 0: H[m,:] = x[m,:] @ W1[:, :300]^T + b1, m in [0, B*T). Row-panel per wave.
__global__ void __launch_bounds__(128) xw1_kernel(const float* __restrict__ x,
                                                  const float* __restrict__ W1,
                                                  const float* __restrict__ b1,
                                                  float* __restrict__ H) {
  const int lane  = threadIdx.x & 31;
  const int wid   = threadIdx.x >> 5;
  const int m0    = (blockIdx.x * 4 + wid) * 16;     // 8192 m-tiles
  const int khalf = lane >> 4;
  v16bf a[KB_];
  load_A_panel(a, x, E_, m0, lane);
#pragma unroll 1
  for (int nt = 0; nt < NT_ - 1; ++nt) {             // full tiles: no predication
    const int n0 = nt * 16;
    v8f acc = tile_accum<false>(a, W1, E2_, 0, n0, lane);
    const int n = n0 + (lane & 15);
    const float bias = b1[n];
#pragma unroll
    for (int r = 0; r < 8; ++r)
      H[(long)(m0 + r + 8 * khalf) * E_ + n] = acc[r] + bias;
  }
  {                                                  // N-tail tile (n0 = 288)
    const int n0 = (NT_ - 1) * 16;
    v8f acc = tile_accum<true>(a, W1, E2_, 0, n0, lane);
    const int n = n0 + (lane & 15);
    if (n < E_) {
      const float bias = b1[n];
#pragma unroll
      for (int r = 0; r < 8; ++r)
        H[(long)(m0 + r + 8 * khalf) * E_ + n] = acc[r] + bias;
    }
  }
}

// Phase 1 (one launch per t): H[b,t,:] = relu(H[b,t,:] + H[b,t-1,:] @ W1[:,300:]^T).
// 16 m-tiles x 19 n-tiles = 304 wave-tiles = 76 blocks of 4 waves.
__global__ void __launch_bounds__(128) step_kernel(float* __restrict__ H,
                                                   const float* __restrict__ W1,
                                                   int t) {
  const int lane  = threadIdx.x & 31;
  const int wid   = threadIdx.x >> 5;
  const int w     = blockIdx.x * 4 + wid;
  const int m0    = (w / NT_) * 16;          // batch tile
  const int nt    = w % NT_;
  const int n0    = nt * 16;
  const int khalf = lane >> 4;
  v8f acc = {};
  if (t > 0) {
    const float* Aprev = H + (long)(t - 1) * E_;   // row index = b, lda = T*E
    v16bf a[KB_];
    load_A_panel(a, Aprev, (long)T_ * E_, m0, lane);
    if (nt < NT_ - 1) acc = tile_accum<false>(a, W1, E2_, E_, n0, lane); // wave-uniform
    else              acc = tile_accum<true >(a, W1, E2_, E_, n0, lane);
  }
  float* slot = H + (long)t * E_;
  const int n = n0 + (lane & 15);
  if (nt < NT_ - 1) {                        // full tile: unmasked epilogue
#pragma unroll
    for (int r = 0; r < 8; ++r) {
      const long idx = (long)(m0 + r + 8 * khalf) * ((long)T_ * E_) + n;
      const float v = slot[idx] + acc[r];
      slot[idx] = v > 0.f ? v : 0.f;
    }
  } else if (n < E_) {
#pragma unroll
    for (int r = 0; r < 8; ++r) {
      const long idx = (long)(m0 + r + 8 * khalf) * ((long)T_ * E_) + n;
      const float v = slot[idx] + acc[r];
      slot[idx] = v > 0.f ? v : 0.f;
    }
  }
}

// h_final = H[:, T-1, :]
__global__ void hfinal_kernel(const float* __restrict__ H, float* __restrict__ hf) {
  const int i = blockIdx.x * blockDim.x + threadIdx.x;
  if (i < B_ * E_) {
    const int b = i / E_, e = i % E_;
    hf[i] = H[((long)b * T_ + (T_ - 1)) * E_ + e];
  }
}

// Phase 2: out[m,:] = H[m,:] @ W2^T + b2, IN-PLACE on H. Wave owns a full 16-row
// panel: loads its entire A panel into registers before any store, so no other
// wave ever reads rows it writes.
__global__ void __launch_bounds__(128) hw2_kernel(float* __restrict__ H,
                                                  const float* __restrict__ W2,
                                                  const float* __restrict__ b2) {
  const int lane  = threadIdx.x & 31;
  const int wid   = threadIdx.x >> 5;
  const int m0    = (blockIdx.x * 4 + wid) * 16;
  const int khalf = lane >> 4;
  v16bf a[KB_];
  load_A_panel(a, H, E_, m0, lane);
#pragma unroll 1
  for (int nt = 0; nt < NT_ - 1; ++nt) {
    const int n0 = nt * 16;
    v8f acc = tile_accum<false>(a, W2, E_, 0, n0, lane);
    const int n = n0 + (lane & 15);
    const float bias = b2[n];
#pragma unroll
    for (int r = 0; r < 8; ++r)
      H[(long)(m0 + r + 8 * khalf) * E_ + n] = acc[r] + bias;
  }
  {
    const int n0 = (NT_ - 1) * 16;
    v8f acc = tile_accum<true>(a, W2, E_, 0, n0, lane);
    const int n = n0 + (lane & 15);
    if (n < E_) {
      const float bias = b2[n];
#pragma unroll
      for (int r = 0; r < 8; ++r)
        H[(long)(m0 + r + 8 * khalf) * E_ + n] = acc[r] + bias;
    }
  }
}

extern "C" void kernel_launch(void* const* d_in, const int* in_sizes, int n_in,
                              void* d_out, int out_size, void* d_ws, size_t ws_size,
                              hipStream_t stream) {
  (void)in_sizes; (void)n_in; (void)out_size; (void)d_ws; (void)ws_size;
  const float* x  = (const float*)d_in[0];
  const float* W1 = (const float*)d_in[1];
  const float* b1 = (const float*)d_in[2];
  const float* W2 = (const float*)d_in[3];
  const float* b2 = (const float*)d_in[4];
  float* out  = (float*)d_out;
  float* H    = out;                                  // [B,T,E] staging = outs slot
  float* hfin = out + (long)B_ * T_ * E_;             // [B,E] tail

  xw1_kernel<<<dim3(2048), dim3(128), 0, stream>>>(x, W1, b1, H);
  for (int t = 0; t < T_; ++t)
    step_kernel<<<dim3(76), dim3(128), 0, stream>>>(H, W1, t);
  hfinal_kernel<<<dim3((B_ * E_ + 255) / 256), dim3(256), 0, stream>>>(H, hfin);
  hw2_kernel<<<dim3(2048), dim3(128), 0, stream>>>(H, W2, b2);
}
